// BitLinear_15058155339886
// MI455X (gfx1250) — matrix operand order repaired and checked by reading, not compiled, not run on hardware
//
#include <hip/hip_runtime.h>

// ---------------------------------------------------------------------------
// BitLinear forward for MI455X (gfx1250, wave32, WMMA + async LDS DMA).
//   y[M,N] = x[M,K] @ dequant(weight)[N,K]^T
//   M = 8192, K = 4096, N = 11008, quant group 128 (along K).
// Pipeline: (1) ternary-quantize weight -> bf16 w_eff in ws
//           (2) convert x -> bf16 in ws
//           (3) bf16 WMMA GEMM, global->LDS staging via
//               GLOBAL_LOAD_ASYNC_TO_LDS_B128 (ASYNCcnt), double buffered.
// ---------------------------------------------------------------------------

#define M_DIM 8192
#define K_DIM 4096
#define N_DIM 11008
#define GROUP 128

#define BM 128
#define BN 256
#define BK 32
#define KT_STEPS (K_DIM / BK)          // 128
#define LDS_STRIDE 40                  // 32 data halfs + 8 pad; 80B rows (16B aligned)
#define ABUF_BYTES (BM * LDS_STRIDE * 2)   // 10240
#define BBUF_BYTES (BN * LDS_STRIDE * 2)   // 20480

typedef __attribute__((ext_vector_type(16))) __bf16 bf16x16;
typedef __attribute__((ext_vector_type(8)))  __bf16 bf16x8;
typedef __attribute__((ext_vector_type(8)))  float  v8f;
typedef __attribute__((ext_vector_type(4)))  float  f32x4;
typedef __attribute__((ext_vector_type(2)))  unsigned int u32x2;

union FragAB { bf16x16 v; bf16x8 h[2]; };

__device__ __forceinline__ unsigned short f32_to_bf16_rne(float f) {
    unsigned int u = __float_as_uint(f);
    unsigned int r = u + 0x7FFFu + ((u >> 16) & 1u);
    return (unsigned short)(r >> 16);
}

// Async global->LDS copy, 16B per lane. VDST holds the LDS byte address,
// VADDR the 64-bit global address. Tracked by ASYNCcnt.
__device__ __forceinline__ void async_load_b128(unsigned lds_addr,
                                                unsigned long long gaddr) {
    asm volatile("global_load_async_to_lds_b128 %0, %1, off"
                 :
                 : "v"(lds_addr), "v"(gaddr)
                 : "memory");
}

__device__ __forceinline__ void wait_async_zero() {
    asm volatile("s_wait_asynccnt 0x0" ::: "memory");
}

// ---------------------------------------------------------------------------
// Kernel 1: per-group abs-mean ternary quantization of weight -> bf16 w_eff.
// One wave32 per group of 128 (4 floats per lane).
// ---------------------------------------------------------------------------
__global__ __launch_bounds__(256) void quant_weight_kernel(
    const float* __restrict__ w, unsigned short* __restrict__ wb, int ngroups) {
    const int group = blockIdx.x * 8 + (threadIdx.x >> 5);
    const int lane  = threadIdx.x & 31;
    if (group >= ngroups) return;

    const f32x4 v = __builtin_nontemporal_load(
        (const f32x4*)(w + (size_t)group * GROUP) + lane);
    float s = fabsf(v.x) + fabsf(v.y) + fabsf(v.z) + fabsf(v.w);
    #pragma unroll
    for (int off = 16; off > 0; off >>= 1) s += __shfl_xor(s, off, 32);

    const float scale = fmaxf(s * (1.0f / 128.0f), 1e-8f);
    const float inv   = 1.0f / scale;

    const float q0 = fminf(fmaxf(rintf(v.x * inv), -1.0f), 1.0f) * scale;
    const float q1 = fminf(fmaxf(rintf(v.y * inv), -1.0f), 1.0f) * scale;
    const float q2 = fminf(fmaxf(rintf(v.z * inv), -1.0f), 1.0f) * scale;
    const float q3 = fminf(fmaxf(rintf(v.w * inv), -1.0f), 1.0f) * scale;

    u32x2 packed;
    packed.x = (unsigned int)f32_to_bf16_rne(q0) | ((unsigned int)f32_to_bf16_rne(q1) << 16);
    packed.y = (unsigned int)f32_to_bf16_rne(q2) | ((unsigned int)f32_to_bf16_rne(q3) << 16);
    *((u32x2*)(wb + (size_t)group * GROUP) + lane) = packed;   // stays cached: re-read by GEMM
}

// ---------------------------------------------------------------------------
// Kernel 2: x f32 -> bf16 (4 elements per thread).
// ---------------------------------------------------------------------------
__global__ __launch_bounds__(256) void cvt_x_kernel(
    const float* __restrict__ x, unsigned short* __restrict__ xb, int n4) {
    const int i = blockIdx.x * blockDim.x + threadIdx.x;
    if (i >= n4) return;
    const f32x4 v = __builtin_nontemporal_load((const f32x4*)x + i);
    u32x2 packed;
    packed.x = (unsigned int)f32_to_bf16_rne(v.x) | ((unsigned int)f32_to_bf16_rne(v.y) << 16);
    packed.y = (unsigned int)f32_to_bf16_rne(v.z) | ((unsigned int)f32_to_bf16_rne(v.w) << 16);
    ((u32x2*)xb)[i] = packed;
}

// ---------------------------------------------------------------------------
// Kernel 3: bf16 WMMA GEMM. C[M,N] = Xb[M,K] * Wb[N,K]^T.
// Block: 256 threads (8 waves), tile 128(M) x 256(N), K-step 32.
// Wave grid 2(M) x 4(N); wave tile 64x64 = 4x4 v_wmma_f32_16x16x32_bf16 accs.
// Staging: async global->LDS DMA (no VGPR staging, no spills), double buffer.
// ---------------------------------------------------------------------------
__global__ __launch_bounds__(256, 1) void bitlinear_wmma_gemm(
    const unsigned short* __restrict__ Xb,   // [M][K] bf16
    const unsigned short* __restrict__ Wb,   // [N][K] bf16
    float* __restrict__ C) {                 // [M][N] f32
    __shared__ unsigned short As[2][BM * LDS_STRIDE];
    __shared__ unsigned short Bs[2][BN * LDS_STRIDE];

    const int tid  = threadIdx.x;
    const int wave = tid >> 5;
    const int lane = tid & 31;
    const int mw   = wave >> 2;   // 0..1
    const int nw   = wave & 3;    // 0..3

    const int m0 = blockIdx.y * BM;
    const int n0 = blockIdx.x * BN;

    v8f acc[4][4] = {};

    // --- per-thread async-copy slots: A = 2 x 16B chunks, B = 4 x 16B chunks
    const unsigned asBase = (unsigned)(size_t)(&As[0][0]);
    const unsigned bsBase = (unsigned)(size_t)(&Bs[0][0]);

    unsigned           laA[2];
    unsigned long long gaA[2];
    #pragma unroll
    for (int i = 0; i < 2; ++i) {
        const int c = tid + i * 256, row = c >> 2, kc = c & 3;
        laA[i] = asBase + (unsigned)(row * LDS_STRIDE + kc * 8) * 2u;
        gaA[i] = (unsigned long long)(size_t)(Xb + (size_t)(m0 + row) * K_DIM + kc * 8);
    }
    unsigned           laB[4];
    unsigned long long gaB[4];
    #pragma unroll
    for (int i = 0; i < 4; ++i) {
        const int c = tid + i * 256, row = c >> 2, kc = c & 3;
        laB[i] = bsBase + (unsigned)(row * LDS_STRIDE + kc * 8) * 2u;
        gaB[i] = (unsigned long long)(size_t)(Wb + (size_t)(n0 + row) * K_DIM + kc * 8);
    }

    // prologue: fill buffer 0, advance global cursors by one K-slice (64B)
    #pragma unroll
    for (int i = 0; i < 2; ++i) { async_load_b128(laA[i], gaA[i]); gaA[i] += 64ull; }
    #pragma unroll
    for (int i = 0; i < 4; ++i) { async_load_b128(laB[i], gaB[i]); gaB[i] += 64ull; }

    const int hi = lane >> 4;   // half-wave select (ISA WMMA lane layout)
    const int r  = lane & 15;

    for (int kt = 0; kt < KT_STEPS; ++kt) {
        wait_async_zero();      // my async writes into buffer (kt&1) are in LDS
        __syncthreads();        // everyone's writes visible; prior readers done

        if (kt + 1 < KT_STEPS) {
            const unsigned nb   = (unsigned)((kt + 1) & 1);
            const unsigned aoff = nb * ABUF_BYTES;
            const unsigned boff = nb * BBUF_BYTES;
            #pragma unroll
            for (int i = 0; i < 2; ++i) { async_load_b128(laA[i] + aoff, gaA[i]); gaA[i] += 64ull; }
            #pragma unroll
            for (int i = 0; i < 4; ++i) { async_load_b128(laB[i] + boff, gaB[i]); gaB[i] += 64ull; }
        }

        const int cur = kt & 1;

        // fragment loads per ISA 7.12.2 bf16 layouts; 4 live B frags, 1 live A frag
        FragAB bfrag[4];
        #pragma unroll
        for (int s = 0; s < 4; ++s) {
            const unsigned short* base = &Bs[cur][(nw * 64 + s * 16 + r) * LDS_STRIDE + hi * 8];
            bfrag[s].h[0] = *reinterpret_cast<const bf16x8*>(base);       // K 0..7  | 8..15
            bfrag[s].h[1] = *reinterpret_cast<const bf16x8*>(base + 16);  // K 16..23| 24..31
        }
        #pragma unroll
        for (int sm = 0; sm < 4; ++sm) {
            FragAB a;
            const unsigned short* base = &As[cur][(mw * 64 + sm * 16 + r) * LDS_STRIDE];
            a.h[0] = *reinterpret_cast<const bf16x8*>(base + hi * 8);        // K 0..7  | 8..15
            a.h[1] = *reinterpret_cast<const bf16x8*>(base + 16 + hi * 8);   // K 16..23| 24..31
            #pragma unroll
            for (int sn = 0; sn < 4; ++sn)
                acc[sm][sn] = __builtin_amdgcn_wmma_f32_16x16x32_bf16(
                    false, a.v, false, bfrag[sn].v,
                    (short)0, acc[sm][sn], false, false);
        }
    }

    // --- epilogue: C/D layout: VGPR v of lane l holds M = v + 8*(l>=16), N = l&15.
    // Nontemporal: output (361 MB) is write-once; keep bf16 operands L2-resident.
    #pragma unroll
    for (int sm = 0; sm < 4; ++sm) {
        #pragma unroll
        for (int sn = 0; sn < 4; ++sn) {
            const int col = n0 + nw * 64 + sn * 16 + r;
            #pragma unroll
            for (int v = 0; v < 8; ++v) {
                const int row = m0 + mw * 64 + sm * 16 + hi * 8 + v;
                __builtin_nontemporal_store(acc[sm][sn][v], &C[(size_t)row * N_DIM + col]);
            }
        }
    }
}

// ---------------------------------------------------------------------------
extern "C" void kernel_launch(void* const* d_in, const int* in_sizes, int n_in,
                              void* d_out, int out_size, void* d_ws, size_t ws_size,
                              hipStream_t stream) {
    const float* x = (const float*)d_in[0];   // [4, 2048, 4096] f32
    const float* w = (const float*)d_in[1];   // [11008, 4096]  f32

    // Workspace layout: Wb bf16 [N][K] (90.2 MB), then Xb bf16 [M][K] (67.1 MB).
    unsigned short* Wb = (unsigned short*)d_ws;
    unsigned short* Xb = Wb + (size_t)N_DIM * K_DIM;

    const int ngroups = (N_DIM * K_DIM) / GROUP;          // 352256
    quant_weight_kernel<<<ngroups / 8, 256, 0, stream>>>(w, Wb, ngroups);

    const int n4 = (M_DIM * K_DIM) / 4;                   // 8388608
    cvt_x_kernel<<<n4 / 256, 256, 0, stream>>>(x, Xb, n4);

    dim3 grid(N_DIM / BN, M_DIM / BM);                    // (43, 64)
    bitlinear_wmma_gemm<<<grid, 256, 0, stream>>>(Xb, Wb, (float*)d_out);
}
